// MultiHeadAttention_46471546143070
// MI455X (gfx1250) — matrix operand order repaired and checked by reading, not compiled, not run on hardware
//
#include <hip/hip_runtime.h>
#include <hip/hip_bf16.h>

typedef __bf16 bf16;
typedef __attribute__((ext_vector_type(16))) __bf16 v16bf;
typedef __attribute__((ext_vector_type(8)))  float  v8f;
typedef __attribute__((ext_vector_type(4)))  int    v4i;

union FragU { v16bf v; unsigned int u[8]; };

// ---------------------------------------------------------------------------
// gfx1250 async global->LDS copy (ASYNCcnt path); fallback: uint4 through VGPRs.
// ---------------------------------------------------------------------------
#ifndef HAVE_ASYNC_LDS
#if defined(__has_builtin)
#if __has_builtin(__builtin_amdgcn_global_load_async_to_lds_b128) && \
    __has_builtin(__builtin_amdgcn_s_wait_asynccnt)
#define HAVE_ASYNC_LDS 1
#else
#define HAVE_ASYNC_LDS 0
#endif
#else
#define HAVE_ASYNC_LDS 0
#endif
#endif

#ifndef HAVE_PERMLANE16
#if defined(__has_builtin)
#if __has_builtin(__builtin_amdgcn_permlane16)
#define HAVE_PERMLANE16 1
#else
#define HAVE_PERMLANE16 0
#endif
#else
#define HAVE_PERMLANE16 0
#endif
#endif

__device__ __forceinline__ void copy16_to_lds(const bf16* g, bf16* l) {
#if HAVE_ASYNC_LDS
  __builtin_amdgcn_global_load_async_to_lds_b128(
      (__attribute__((address_space(1))) v4i*)g,
      (__attribute__((address_space(3))) v4i*)l, 0, 0);
#else
  *(uint4*)l = *(const uint4*)g;
#endif
}

__device__ __forceinline__ void copy_join() {
#if HAVE_ASYNC_LDS
  __builtin_amdgcn_s_wait_asynccnt(0);
#endif
}

// max-reduce across the 16 lanes of a half-wave (VALU v_permlane16, no LDS)
__device__ __forceinline__ float half_xor_max(float v) {
#if HAVE_PERMLANE16
  int x = __builtin_bit_cast(int, v), y;
  y = __builtin_amdgcn_permlane16(x, x, 0x67452301u, 0xEFCDAB89u, false, false);
  v = fmaxf(v, __builtin_bit_cast(float, y)); x = __builtin_bit_cast(int, v);
  y = __builtin_amdgcn_permlane16(x, x, 0x54761032u, 0xDCFE98BAu, false, false);
  v = fmaxf(v, __builtin_bit_cast(float, y)); x = __builtin_bit_cast(int, v);
  y = __builtin_amdgcn_permlane16(x, x, 0x32107654u, 0xBA98FEDCu, false, false);
  v = fmaxf(v, __builtin_bit_cast(float, y)); x = __builtin_bit_cast(int, v);
  y = __builtin_amdgcn_permlane16(x, x, 0xFEDCBA98u, 0x76543210u, false, false);
  v = fmaxf(v, __builtin_bit_cast(float, y));
#else
#pragma unroll
  for (int m = 1; m < 16; m <<= 1)
    v = fmaxf(v, __shfl_xor(v, m, 32));
#endif
  return v;
}

// ---------------------------------------------------------------------------
// WMMA fragment helpers (CDNA5 16-bit layouts, wave32).
// A (16x32, MxK): lane m=lane&15, half lane -> +8 in K per pair group.
// B (32x16, KxN): lane n=lane&15, half lane -> +8 in K per pair group;
//   tiles staged in LDS as [n][k] so each fragment is 2x ds_load_b128.
// C/D (16x16 f32): VGPR i, lane l -> (M = i + 8*(l>>4), N = l&15).
// Row strides kept at multiples of 8 elements (16 B) so b128 merging works.
// ---------------------------------------------------------------------------
__device__ __forceinline__ v16bf load_frag_a(const bf16* base, int stride, int lane) {
  const int m     = lane & 15;
  const int half8 = (lane & 16) >> 1;
  const unsigned short* p = (const unsigned short*)(base + m * stride);
  FragU f;
#pragma unroll
  for (int i = 0; i < 4; ++i) {
    f.u[i]     = *(const unsigned int*)(p + half8 + 2 * i);
    f.u[i + 4] = *(const unsigned int*)(p + 16 + half8 + 2 * i);
  }
  return f.v;
}

__device__ __forceinline__ v16bf load_frag_b(const bf16* base, int stride, int lane) {
  const int n      = lane & 15;
  const int half8  = (lane & 16) >> 1;
  const unsigned short* p = (const unsigned short*)(base + n * stride);
  FragU f;
#pragma unroll
  for (int i = 0; i < 4; ++i) {
    f.u[i]     = *(const unsigned int*)(p + half8 + 2 * i);
    f.u[i + 4] = *(const unsigned int*)(p + 16 + half8 + 2 * i);
  }
  return f.v;
}

__device__ __forceinline__ v16bf ones_frag() {
  FragU f;
#pragma unroll
  for (int i = 0; i < 8; ++i) f.u[i] = 0x3F803F80u;  // two bf16 1.0
  return f.v;
}

__device__ __forceinline__ v8f wmma_bf16(v16bf a, v16bf b, v8f c) {
  return __builtin_amdgcn_wmma_f32_16x16x32_bf16(false, a, false, b,
                                                 (short)0, c, false, false);
}

// ---------------------------------------------------------------------------
// fp32 -> bf16 conversion (4 elements/thread)
// ---------------------------------------------------------------------------
__global__ __launch_bounds__(256) void convert_kernel(const float* __restrict__ src,
                                                      bf16* __restrict__ dst, int n) {
  int i = (blockIdx.x * 256 + threadIdx.x) * 4;
  if (i < n) {
    float4 v = *(const float4*)(src + i);
    dst[i + 0] = (bf16)v.x;
    dst[i + 1] = (bf16)v.y;
    dst[i + 2] = (bf16)v.z;
    dst[i + 3] = (bf16)v.w;
  }
}

// ---------------------------------------------------------------------------
// Transpose + convert: Wt_bf16[n][k] = (bf16)W_f32[k][n], 1024x1024.
// ---------------------------------------------------------------------------
__global__ __launch_bounds__(256) void transpose_convert_kernel(
    const float* __restrict__ src, bf16* __restrict__ dst) {
  __shared__ float t[64][65];
  const int kBase = blockIdx.x * 64, nBase = blockIdx.y * 64;
  const int r = threadIdx.x >> 2, c = (threadIdx.x & 3) * 16;
  const float4* s4 = (const float4*)(src + (size_t)(kBase + r) * 1024 + nBase + c);
#pragma unroll
  for (int j = 0; j < 4; ++j) {
    float4 v = s4[j];
    t[r][c + j * 4 + 0] = v.x;
    t[r][c + j * 4 + 1] = v.y;
    t[r][c + j * 4 + 2] = v.z;
    t[r][c + j * 4 + 3] = v.w;
  }
  __syncthreads();
  bf16 out[16];
#pragma unroll
  for (int j = 0; j < 16; ++j) out[j] = (bf16)t[c + j][r];
  bf16* d = dst + (size_t)(nBase + r) * 1024 + kBase + c;
  ((uint4*)d)[0] = ((uint4*)out)[0];
  ((uint4*)d)[1] = ((uint4*)out)[1];
}

// ---------------------------------------------------------------------------
// Tiled WMMA GEMM: C[8192,1024] = A_bf16[8192,1024] @ W (given as Wt[n][k]).
// Block tile 128x128, BK=64, double-buffered async LDS staging.
// MODE: 0 = bf16 row-major out, 1 = f32 row-major out, 2 = bf16 transposed.
// ---------------------------------------------------------------------------
template <int MODE>
__global__ __launch_bounds__(256) void gemm_kernel(const bf16* __restrict__ A,
                                                   const bf16* __restrict__ Wt,
                                                   float* __restrict__ Cf,
                                                   bf16* __restrict__ Cb) {
  constexpr int K = 1024, N = 1024, M = 8192;
  constexpr int BM = 128, BN = 128, BK = 64;
  constexpr int AS = BK + 8;
  constexpr int BS = BK + 8;
  __shared__ bf16 As[2][BM * AS];
  __shared__ bf16 Bs[2][BN * BS];

  const int tid  = threadIdx.x;
  const int lane = tid & 31;
  const int wave = tid >> 5;
  const int wm = wave >> 1;
  const int wn = wave & 1;
  const int mBase = blockIdx.x * BM;
  const int nBase = blockIdx.y * BN;

  const int r  = tid >> 1;
  const int cc = (tid & 1) * 32;

  const bf16* agBase = A  + (size_t)(mBase + r) * K + cc;
  const bf16* bgBase = Wt + (size_t)(nBase + r) * K + cc;

  auto stage = [&](int buf, int k0) {
    const bf16* ag = agBase + k0;
    const bf16* bg = bgBase + k0;
    bf16* al = &As[buf][r * AS + cc];
    bf16* bl = &Bs[buf][r * BS + cc];
#pragma unroll
    for (int j = 0; j < 4; ++j) copy16_to_lds(ag + j * 8, al + j * 8);
#pragma unroll
    for (int j = 0; j < 4; ++j) copy16_to_lds(bg + j * 8, bl + j * 8);
  };

  v8f acc[2][4] = {};

  stage(0, 0);
  copy_join();
  __syncthreads();

  int buf = 0;
  for (int k0 = 0; k0 < K; k0 += BK) {
    if (k0 + BK < K) stage(buf ^ 1, k0 + BK);
    if (k0 + 2 * BK < K) {
      __builtin_prefetch(agBase + k0 + 2 * BK, 0, 1);
      __builtin_prefetch(bgBase + k0 + 2 * BK, 0, 1);
    }

    const bf16* asb = As[buf];
    const bf16* bsb = Bs[buf];
#pragma unroll
    for (int kb = 0; kb < 2; ++kb) {
      v16bf a0 = load_frag_a(&asb[(wm * 32 + 0)  * AS + kb * 32], AS, lane);
      v16bf a1 = load_frag_a(&asb[(wm * 32 + 16) * AS + kb * 32], AS, lane);
#pragma unroll
      for (int ni = 0; ni < 4; ++ni) {
        v16bf bw = load_frag_b(&bsb[(wn * 64 + ni * 16) * BS + kb * 32], BS, lane);
        acc[0][ni] = wmma_bf16(a0, bw, acc[0][ni]);
        acc[1][ni] = wmma_bf16(a1, bw, acc[1][ni]);
      }
    }
    copy_join();
    __syncthreads();
    buf ^= 1;
  }

  const int nl    = lane & 15;
  const int half8 = (lane & 16) >> 1;
#pragma unroll
  for (int mi = 0; mi < 2; ++mi)
#pragma unroll
    for (int ni = 0; ni < 4; ++ni)
#pragma unroll
      for (int i = 0; i < 8; ++i) {
        const size_t row = (size_t)mBase + wm * 32 + mi * 16 + i + half8;
        const size_t col = (size_t)nBase + wn * 64 + ni * 16 + nl;
        if constexpr (MODE == 1)      Cf[row * N + col] = acc[mi][ni][i];
        else if constexpr (MODE == 0) Cb[row * N + col] = (bf16)acc[mi][ni][i];
        else                          Cb[col * M + row] = (bf16)acc[mi][ni][i];
      }
}

// ---------------------------------------------------------------------------
// Flash attention (causal, online softmax), bf16 WMMA, f32 accum.
// Grid: (S/64, B*H). Block: 128 threads = 4 waves; each wave owns 16 q rows.
// Row sums accumulated as a 5th WMMA accumulator (P @ ones) so the only
// cross-lane reduction left is the row max (v_permlane16). Double-buffered
// async K/V staging; one barrier per kv tile.
// ---------------------------------------------------------------------------
__global__ __launch_bounds__(128) void attn_kernel(const bf16* __restrict__ Qb,
                                                   const bf16* __restrict__ Kb,
                                                   const bf16* __restrict__ Vt,
                                                   bf16* __restrict__ Ctx) {
  constexpr int S = 2048, DM = 1024, DK = 64, M = 8192;
  constexpr int QT = 64, KT = 64;
  constexpr int KS = DK + 8;
  constexpr int VS = KT + 8;
  constexpr int PS = KT + 8;
  __shared__ bf16 Kt[2][KT * KS];            // [kv][d]
  __shared__ bf16 Vx[2][DK * VS];            // [d][kv]
  __shared__ bf16 Pl[4][16 * PS];

  const int tid  = threadIdx.x;
  const int lane = tid & 31;
  const int wave = tid >> 5;
  const int bh = blockIdx.y;
  const int b = bh >> 4, h = bh & 15;
  const int qBase = blockIdx.x * QT;
  const int qRow0 = qBase + wave * 16;
  const size_t headOff = (size_t)b * S * DM + (size_t)h * DK;
  const bf16* vthead = Vt + (size_t)(h * DK) * M + (size_t)b * S;

  const int nl    = lane & 15;
  const int half8 = (lane & 16) >> 1;
  const int r  = tid >> 1;
  const int cc = (tid & 1) * 32;

  auto stage = [&](int buf, int kv0) {
    const bf16* kg = Kb + headOff + (size_t)(kv0 + r) * DM + cc;
    const bf16* vg = vthead + (size_t)r * M + kv0 + cc;
    bf16* kl = &Kt[buf][r * KS + cc];
    bf16* vl = &Vx[buf][r * VS + cc];
#pragma unroll
    for (int j = 0; j < 4; ++j) copy16_to_lds(kg + j * 8, kl + j * 8);
#pragma unroll
    for (int j = 0; j < 4; ++j) copy16_to_lds(vg + j * 8, vl + j * 8);
  };

  // Q fragments (A layout), two 32-wide k-blocks covering d_k = 64
  v16bf qf[2];
  {
    const unsigned short* p =
        (const unsigned short*)(Qb + headOff + (size_t)(qRow0 + nl) * DM);
#pragma unroll
    for (int kb = 0; kb < 2; ++kb) {
      FragU f;
#pragma unroll
      for (int i = 0; i < 4; ++i) {
        f.u[i]     = *(const unsigned int*)(p + kb * 32 + half8 + 2 * i);
        f.u[i + 4] = *(const unsigned int*)(p + kb * 32 + 16 + half8 + 2 * i);
      }
      qf[kb] = f.v;
    }
  }

  const v16bf ones = ones_frag();

  v8f ctx[4] = {};
  v8f csum = {};                             // row sums via P @ ones
  float mrow[8];
#pragma unroll
  for (int i = 0; i < 8; ++i) mrow[i] = -__builtin_inff();

  const float scale = 0.125f;                // 1/sqrt(64)
  const int kvEnd = qBase + QT;

  stage(0, 0);
  copy_join();
  __syncthreads();

  int buf = 0;
  for (int kv0 = 0; kv0 < kvEnd; kv0 += KT) {
    if (kv0 + KT < kvEnd) stage(buf ^ 1, kv0 + KT);  // async: overlaps compute

    // S = Q K^T (per-wave 16 x 64 tile)
    const bf16* kt = Kt[buf];
    v8f sf[4];
#pragma unroll
    for (int ni = 0; ni < 4; ++ni) {
      v8f z = {};
      v16bf b0 = load_frag_b(&kt[(ni * 16) * KS + 0],  KS, lane);
      v16bf b1 = load_frag_b(&kt[(ni * 16) * KS + 32], KS, lane);
      z = wmma_bf16(qf[0], b0, z);
      z = wmma_bf16(qf[1], b1, z);
      sf[ni] = z;
    }

    // scale + causal mask + per-row max
    float rmax[8];
#pragma unroll
    for (int i = 0; i < 8; ++i) rmax[i] = -__builtin_inff();
#pragma unroll
    for (int ni = 0; ni < 4; ++ni) {
      const int kvg = kv0 + ni * 16 + nl;
#pragma unroll
      for (int i = 0; i < 8; ++i) {
        const int qg = qRow0 + i + half8;
        float v = sf[ni][i] * scale;
        v = (kvg <= qg) ? v : -__builtin_inff();
        sf[ni][i] = v;
        rmax[i] = fmaxf(rmax[i], v);
      }
    }
#pragma unroll
    for (int i = 0; i < 8; ++i) rmax[i] = half_xor_max(rmax[i]);

    // online softmax update (row sum handled by the P @ ones accumulator)
#pragma unroll
    for (int i = 0; i < 8; ++i) {
      const float mn = fmaxf(mrow[i], rmax[i]);
      const float alpha = __expf(mrow[i] - mn);
      mrow[i] = mn;
#pragma unroll
      for (int ni = 0; ni < 4; ++ni)
        sf[ni][i] = __expf(sf[ni][i] - mn);
#pragma unroll
      for (int ni = 0; ni < 4; ++ni) ctx[ni][i] *= alpha;
      csum[i] *= alpha;
    }

    // P: C-layout -> A-layout via this wave's private LDS patch
    bf16* pw = Pl[wave];
#pragma unroll
    for (int ni = 0; ni < 4; ++ni)
#pragma unroll
      for (int i = 0; i < 8; ++i)
        pw[(i + half8) * PS + ni * 16 + nl] = (bf16)sf[ni][i];

    // ctx += P @ V ; csum += P @ ones
    const bf16* vx = Vx[buf];
#pragma unroll
    for (int kb = 0; kb < 2; ++kb) {
      v16bf pa = load_frag_a(&pw[kb * 32], PS, lane);
      csum = wmma_bf16(pa, ones, csum);
#pragma unroll
      for (int ni = 0; ni < 4; ++ni) {
        v16bf vb = load_frag_b(&vx[(ni * 16) * VS + kb * 32], VS, lane);
        ctx[ni] = wmma_bf16(pa, vb, ctx[ni]);
      }
    }
    copy_join();
    __syncthreads();
    buf ^= 1;
  }

  // normalize and store context (bf16) back in [B*S, D_MODEL] layout
#pragma unroll
  for (int i = 0; i < 8; ++i) {
    const float inv = 1.0f / csum[i];
    const size_t row = (size_t)(qRow0 + i + half8);
#pragma unroll
    for (int ni = 0; ni < 4; ++ni)
      Ctx[headOff + row * DM + ni * 16 + nl] = (bf16)(ctx[ni][i] * inv);
  }
}

// ---------------------------------------------------------------------------
// Launcher
// ---------------------------------------------------------------------------
extern "C" void kernel_launch(void* const* d_in, const int* in_sizes, int n_in,
                              void* d_out, int out_size, void* d_ws, size_t ws_size,
                              hipStream_t stream) {
  const float* x  = (const float*)d_in[0];
  const float* Wq = (const float*)d_in[1];
  const float* Wk = (const float*)d_in[2];
  const float* Wv = (const float*)d_in[3];
  const float* Wo = (const float*)d_in[4];

  constexpr size_t XN = (size_t)8192 * 1024;
  constexpr size_t WN = (size_t)1024 * 1024;

  bf16* xb  = (bf16*)d_ws;
  bf16* wqt = xb + XN;
  bf16* wkt = wqt + WN;
  bf16* wvt = wkt + WN;
  bf16* wot = wvt + WN;
  bf16* Qb  = wot + WN;
  bf16* Kb  = Qb + XN;
  bf16* Vtb = Kb + XN;   // V^T: [1024 = h*64+d][8192 = b*2048+s]
  bf16* Cx  = Vtb + XN;

  convert_kernel<<<XN / 1024, 256, 0, stream>>>(x, xb, (int)XN);
  dim3 tg(16, 16);
  transpose_convert_kernel<<<tg, 256, 0, stream>>>(Wq, wqt);
  transpose_convert_kernel<<<tg, 256, 0, stream>>>(Wk, wkt);
  transpose_convert_kernel<<<tg, 256, 0, stream>>>(Wv, wvt);
  transpose_convert_kernel<<<tg, 256, 0, stream>>>(Wo, wot);

  dim3 gg(64, 8);
  gemm_kernel<0><<<gg, 256, 0, stream>>>(xb, wqt, nullptr, Qb);
  gemm_kernel<0><<<gg, 256, 0, stream>>>(xb, wkt, nullptr, Kb);
  gemm_kernel<2><<<gg, 256, 0, stream>>>(xb, wvt, nullptr, Vtb);

  attn_kernel<<<dim3(32, 64), 128, 0, stream>>>(Qb, Kb, Vtb, Cx);

  gemm_kernel<1><<<gg, 256, 0, stream>>>(Cx, wot, (float*)d_out, nullptr);
}
